// MultiHeadAttention_48395691492005
// MI455X (gfx1250) — compile-verified, matrix operand order checked by reading, not run on hardware
//
#include <hip/hip_runtime.h>
#include <hip/hip_bf16.h>

typedef __attribute__((ext_vector_type(16))) __bf16 v16bf;
typedef __attribute__((ext_vector_type(8)))  __bf16 v8bf;
typedef __attribute__((ext_vector_type(4)))  __bf16 v4bf;
typedef __attribute__((ext_vector_type(8)))  float  v8f;
typedef __attribute__((ext_vector_type(4)))  float  v4f;
typedef __attribute__((ext_vector_type(4)))  unsigned int u32x4;
typedef __attribute__((ext_vector_type(8)))  int i32x8;
typedef __attribute__((ext_vector_type(4)))  int i32x4;

#define D_MODEL 1024
#define SEQ     2048
#define BATCH   4
#define HEADS   16
#define DKH     64
#define M_TOT   (BATCH * SEQ)   // 8192

// GEMM tiling
#define TM 128
#define TN 128
#define TK 32
#define LDP (TK + 8)            // LDS row pitch (bf16): 80B = TDM row 64B + 16B pad

#define HAVE_TDM __has_builtin(__builtin_amdgcn_tensor_load_to_lds)

// group1 word0 for padded bf16 tile loads:
// data_size=1 (2B) | pad_enable | pad_interval=3 (16 DWORDs=64B) | pad_amount=3 (4 DWORDs=16B)
#define G1W0_PAD  (0x00010000u | (1u << 20) | (3u << 22) | (3u << 25))
#define G1W0_NOPAD 0x00010000u

__device__ __forceinline__ v8f wmma_bf16(v16bf a, v16bf b, v8f c) {
    // D = A(16x32 bf16) * B(32x16 bf16) + C(16x16 f32)
    return __builtin_amdgcn_wmma_f32_16x16x32_bf16(
        /*neg_a=*/false, a, /*neg_b=*/false, b,
        /*c_mod=*/(short)0, c, /*reuse_a=*/false, /*reuse_b=*/false);
}

// Load a 16-element bf16 fragment as two 16B chunks (ds_load_b128 / global_load_b128 x2)
__device__ __forceinline__ v16bf load_frag16(const __bf16* p) {
    v8bf lo = *(const v8bf*)p;
    v8bf hi = *(const v8bf*)(p + 8);
    v16bf r;
#pragma unroll
    for (int i = 0; i < 8; ++i) { r[i] = lo[i]; r[i + 8] = hi[i]; }
    return r;
}

__device__ __forceinline__ v4bf cvt4(v4f f) {
    v4bf r;
    r[0] = (__bf16)f[0]; r[1] = (__bf16)f[1];
    r[2] = (__bf16)f[2]; r[3] = (__bf16)f[3];
    return r;
}

// ---- Tensor Data Mover: 2-D tile load, descriptor built per ISA 08_async_tensor §8 ----
// group0: [1:0]=count=1, [63:32]=lds_addr, [120:64]=global_addr, [127:126]=type=2
// group1: [17:16]=data_size, [20]=pad_en, [24:22]=pad_interval, [31:25]=pad_amount,
//         [79:48]=tensor_dim0, [111:80]=tensor_dim1, [127:112]=tile_dim0,
//         [143:128]=tile_dim1, [207:160]=tensor_dim0_stride
__device__ __forceinline__ void tdm_load_2d(unsigned lds_addr, const void* gptr, i32x8 g1) {
#if HAVE_TDM
    unsigned long long ga = (unsigned long long)gptr;
    u32x4 g0;
    g0[0] = 1u;                                   // count = 1, user mode
    g0[1] = lds_addr;                             // LDS byte address
    g0[2] = (unsigned)ga;                         // global_addr[31:0]
    g0[3] = (unsigned)(ga >> 32) | (2u << 30);    // global_addr[56:32] | type=2
    i32x4 z4 = {0, 0, 0, 0};
#if __has_include(<hip/amd_detail/amd_gfx1250_TDM.h>)
    i32x8 z8 = {0, 0, 0, 0, 0, 0, 0, 0};
    __builtin_amdgcn_tensor_load_to_lds(g0, g1, z4, z4, z8, 0);
#else
    __builtin_amdgcn_tensor_load_to_lds(g0, g1, z4, z4, 0);
#endif
#endif
}

// Elementwise fp32 -> bf16 (vectorized x4); one-time pass so GEMM tiles are DMA-able bf16.
__global__ __launch_bounds__(256)
void cvt_f32_bf16(const float* __restrict__ in, __bf16* __restrict__ out, int n4)
{
    const int i = blockIdx.x * blockDim.x + threadIdx.x;
    if (i < n4) ((v4bf*)out)[i] = cvt4(((const v4f*)in)[i]);
}

// C[M,N] = A[M,K] * W[N,K]^T + bias, M=8192, N=K=1024, A/W bf16.
// Tiles staged into LDS by the TDM (wave 0 issues, double-buffered, padded rows).
// MODE 0: bf16 out, [B,H,S,Dk] layout (Q/K proj)
// MODE 2: bf16 out, [B,H,Dk,S] layout (V proj, transposed for PV B-fragments)
// MODE 3: fp32 out, [M,N] plain (final O proj)
template <int MODE>
__global__ __launch_bounds__(256)
void gemm_wmma(const __bf16* __restrict__ A16, const __bf16* __restrict__ W16,
               const float* __restrict__ bias, void* __restrict__ out)
{
    __shared__ __bf16 As[2][TM][LDP];
    __shared__ __bf16 Bs[2][TN][LDP];

    const int tid  = threadIdx.x;
    const int lane = tid & 31;
    const int wave = tid >> 5;
    const int wm   = wave & 3;      // 4 waves down M: 32 rows each
    const int wn   = wave >> 2;     // 2 waves across N: 64 cols each
    const int m0   = blockIdx.y * TM;
    const int n0   = blockIdx.x * TN;

    const v8f z = {0.f, 0.f, 0.f, 0.f, 0.f, 0.f, 0.f, 0.f};
    v8f acc[2][4];
#pragma unroll
    for (int mt = 0; mt < 2; ++mt)
#pragma unroll
        for (int nt = 0; nt < 4; ++nt) acc[mt][nt] = z;

    const bool issuer = (wave == 0);
    // 128x32 bf16 tile from a [rows x 1024] row-major tensor, padded to 80B LDS rows
    const i32x8 g1a = {(int)G1W0_PAD, 1024 << 16, 8192 << 16, 32 << 16, 128, 1024, 0, 0};
    const i32x8 g1w = {(int)G1W0_PAD, 1024 << 16, 1024 << 16, 32 << 16, 128, 1024, 0, 0};
    auto issue = [&](int buf, int kb) {
        const int k0 = kb * TK;
        tdm_load_2d((unsigned)(size_t)&As[buf][0][0], A16 + (size_t)m0 * D_MODEL + k0, g1a);
        tdm_load_2d((unsigned)(size_t)&Bs[buf][0][0], W16 + (size_t)n0 * D_MODEL + k0, g1w);
    };

#if !HAVE_TDM
    // Fallback: cooperative bf16 staging, 256 threads x 16 elems per tile
    auto stage = [&](int buf, int kb) {
        const int k0  = kb * TK;
        const int idx = tid * 16;
        const int row = idx >> 5, col = idx & 31;   // col in {0,16}
        *(v8bf*)&As[buf][row][col]     = *(const v8bf*)&A16[(size_t)(m0 + row) * D_MODEL + k0 + col];
        *(v8bf*)&As[buf][row][col + 8] = *(const v8bf*)&A16[(size_t)(m0 + row) * D_MODEL + k0 + col + 8];
        *(v8bf*)&Bs[buf][row][col]     = *(const v8bf*)&W16[(size_t)(n0 + row) * D_MODEL + k0 + col];
        *(v8bf*)&Bs[buf][row][col + 8] = *(const v8bf*)&W16[(size_t)(n0 + row) * D_MODEL + k0 + col + 8];
    };
    stage(0, 0);
#else
    if (issuer) issue(0, 0);
#endif

    const int rsel = lane & 15;
    const int koff = (lane < 16) ? 0 : 16;   // contiguous K halves (k = 16h + s), same order for A and B
    const int NKB  = D_MODEL / TK;           // 32

    for (int kb = 0; kb < NKB; ++kb) {
        const int buf = kb & 1;
#if HAVE_TDM
        if (issuer) __builtin_amdgcn_s_wait_tensorcnt(0);   // buf landed in LDS
        __syncthreads();                                    // publish buf
        if (issuer && kb + 1 < NKB) issue(buf ^ 1, kb + 1); // overlap next tiles
#else
        __syncthreads();
#endif

        v16bf af0 = load_frag16(&As[buf][wm * 32 + rsel][koff]);
        v16bf af1 = load_frag16(&As[buf][wm * 32 + 16 + rsel][koff]);
#pragma unroll
        for (int nt = 0; nt < 4; ++nt) {
            v16bf bf = load_frag16(&Bs[buf][wn * 64 + nt * 16 + rsel][koff]);
            acc[0][nt] = wmma_bf16(af0, bf, acc[0][nt]);
            acc[1][nt] = wmma_bf16(af1, bf, acc[1][nt]);
        }
        __syncthreads();   // compute on buf done before it is refilled
#if !HAVE_TDM
        if (kb + 1 < NKB) stage(buf ^ 1, kb + 1);
#endif
    }

    // Epilogue: C layout = lane n (0..15 twice), VGPR r -> M = r (+8 for lanes 16..31)
    const int rowadd = (lane >> 4) * 8;
    const int nlane  = lane & 15;
#pragma unroll
    for (int mt = 0; mt < 2; ++mt)
#pragma unroll
        for (int nt = 0; nt < 4; ++nt)
#pragma unroll
            for (int r = 0; r < 8; ++r) {
                const int m = m0 + wm * 32 + mt * 16 + rowadd + r;
                const int n = n0 + wn * 64 + nt * 16 + nlane;
                const float val = acc[mt][nt][r] + bias[n];
                if (MODE == 0) {
                    const int b = m >> 11, s = m & 2047, h = n >> 6, dk = n & 63;
                    ((__bf16*)out)[((((size_t)b * HEADS + h) * SEQ + s) << 6) + dk] = (__bf16)val;
                } else if (MODE == 2) {
                    const int b = m >> 11, s = m & 2047, h = n >> 6, dk = n & 63;
                    ((__bf16*)out)[(((size_t)b * HEADS + h) * DKH + dk) * SEQ + s] = (__bf16)val;
                } else {
                    ((float*)out)[(size_t)m * D_MODEL + n] = val;
                }
            }
}

// Flash attention: grid (S/64, B*H), 128 threads (4 waves), each wave owns 16 q rows.
// K/V tiles staged into LDS by the TDM (wave 0 issues, TENSORcnt + barrier publish),
// double-buffered so the DMA overlaps WMMA/softmax work.
// q,k: bf16 [B,H,S,Dk]; vT: bf16 [B,H,Dk,S]; out: bf16 [B,S,D] (heads re-concatenated).
__global__ __launch_bounds__(128)
void flash_attn(const __bf16* __restrict__ q, const __bf16* __restrict__ k,
                const __bf16* __restrict__ vT, __bf16* __restrict__ outp)
{
    __shared__ __bf16 Kt[2][32][64];   // KV-step tile of K: 32 kv rows x 64 dk (8 KB)
    __shared__ __bf16 Vt[2][64][32];   // KV-step tile of V^T: 64 dk rows x 32 kv (8 KB)
    __shared__ __bf16 Ps[4][16][40];   // per-wave P tile (16 x 32), padded rows

    const int lane = threadIdx.x & 31;
    const int wave = threadIdx.x >> 5;
    const int bh   = blockIdx.y;
    const int b    = bh >> 4, h = bh & 15;
    const int q0   = blockIdx.x * 64 + wave * 16;

    const __bf16* qp = q  + (size_t)bh * SEQ * DKH;
    const __bf16* kp = k  + (size_t)bh * SEQ * DKH;
    const __bf16* vp = vT + (size_t)bh * DKH * SEQ;

    const int rsel   = lane & 15;
    const int koff   = (lane < 16) ? 0 : 16;
    const int rowadd = (lane >> 4) * 8;

    // Q fragments for K = 0..31 and 32..63 (loaded once, reused all iterations)
    const v16bf qf0 = load_frag16(&qp[(size_t)(q0 + rsel) * DKH + koff]);
    const v16bf qf1 = load_frag16(&qp[(size_t)(q0 + rsel) * DKH + 32 + koff]);

    const v8f z = {0.f, 0.f, 0.f, 0.f, 0.f, 0.f, 0.f, 0.f};
    v8f acc[4];
#pragma unroll
    for (int nt = 0; nt < 4; ++nt) acc[nt] = z;

    float m_r[8], l_r[8];
#pragma unroll
    for (int r = 0; r < 8; ++r) { m_r[r] = -3.0e38f; l_r[r] = 0.f; }

    const float scale = 0.125f;   // 1/sqrt(64)
    const int   NSTEP = SEQ / 32; // 64

    // K tile: tensor 64 x SEQ (dim0 contiguous = dk), tile 64 x 32, stride0 = 64
    const i32x8 g1k = {(int)G1W0_NOPAD, 64 << 16, 2048 << 16, 64 << 16, 32, 64, 0, 0};
    // V tile: tensor SEQ x 64 (dim0 contiguous = kv), tile 32 x 64, stride0 = SEQ
    const i32x8 g1v = {(int)G1W0_NOPAD, 2048 << 16, 64 << 16, 32 << 16, 64, 2048, 0, 0};

    const bool issuer = (wave == 0);
    auto issue = [&](int buf, int kv) {
        tdm_load_2d((unsigned)(size_t)&Kt[buf][0][0], kp + (size_t)kv * DKH, g1k);
        tdm_load_2d((unsigned)(size_t)&Vt[buf][0][0], vp + kv, g1v);
    };

#if !HAVE_TDM
    // Fallback: cooperative global->LDS staging (128 threads, 16 bf16 each)
    auto stage = [&](int buf, int kv) {
        const int idx = threadIdx.x * 16;
        {   // K tile: flat idx -> [row 0..31][col 0..63]
            const int row = idx >> 6, col = idx & 63;
            *(v8bf*)&Kt[buf][row][col]     = *(const v8bf*)&kp[(size_t)(kv + row) * DKH + col];
            *(v8bf*)&Kt[buf][row][col + 8] = *(const v8bf*)&kp[(size_t)(kv + row) * DKH + col + 8];
        }
        {   // V tile: flat idx -> [row 0..63][col 0..31]
            const int row = idx >> 5, col = idx & 31;
            *(v8bf*)&Vt[buf][row][col]     = *(const v8bf*)&vp[(size_t)row * SEQ + kv + col];
            *(v8bf*)&Vt[buf][row][col + 8] = *(const v8bf*)&vp[(size_t)row * SEQ + kv + col + 8];
        }
    };
    stage(0, 0);
#else
    if (issuer) issue(0, 0);
#endif

    for (int i = 0; i < NSTEP; ++i) {
        const int buf = i & 1;
        const int kv  = i * 32;

#if HAVE_TDM
        if (issuer) __builtin_amdgcn_s_wait_tensorcnt(0);      // buf landed in LDS
        __syncthreads();                                       // publish buf
        if (issuer && i + 1 < NSTEP) issue(buf ^ 1, kv + 32);  // overlap next tile
#else
        __syncthreads();
#endif

        // ---- scores: two 16x16 tiles over K=64 (two WMMA K-steps each) ----
        v8f s0 = z, s1 = z;
        {
            v16bf kf;
            kf = load_frag16(&Kt[buf][rsel][koff]);
            s0 = wmma_bf16(qf0, kf, s0);
            kf = load_frag16(&Kt[buf][rsel][32 + koff]);
            s0 = wmma_bf16(qf1, kf, s0);
            kf = load_frag16(&Kt[buf][16 + rsel][koff]);
            s1 = wmma_bf16(qf0, kf, s1);
            kf = load_frag16(&Kt[buf][16 + rsel][32 + koff]);
            s1 = wmma_bf16(qf1, kf, s1);
        }

        // ---- online softmax (reductions stay inside each 16-lane half) ----
        float alpha[8];
#pragma unroll
        for (int r = 0; r < 8; ++r) {
            float a = s0[r] * scale;
            float c = s1[r] * scale;
            float mx = fmaxf(a, c);
#pragma unroll
            for (int off = 1; off < 16; off <<= 1)
                mx = fmaxf(mx, __shfl_xor(mx, off, 32));
            const float mnew = fmaxf(m_r[r], mx);
            const float p0 = __expf(a - mnew);
            const float p1 = __expf(c - mnew);
            float rs = p0 + p1;
#pragma unroll
            for (int off = 1; off < 16; off <<= 1)
                rs += __shfl_xor(rs, off, 32);
            alpha[r] = __expf(m_r[r] - mnew);
            l_r[r]   = l_r[r] * alpha[r] + rs;
            m_r[r]   = mnew;
            s0[r] = p0; s1[r] = p1;
        }
#pragma unroll
        for (int nt = 0; nt < 4; ++nt)
#pragma unroll
            for (int r = 0; r < 8; ++r) acc[nt][r] *= alpha[r];

        // ---- re-fragment P (C layout -> A layout) through per-wave LDS ----
#pragma unroll
        for (int r = 0; r < 8; ++r) {
            Ps[wave][rowadd + r][rsel]      = (__bf16)s0[r];
            Ps[wave][rowadd + r][16 + rsel] = (__bf16)s1[r];
        }
        asm volatile("s_wait_dscnt 0" ::: "memory");  // per-wave LDS RAW
        const v16bf pf = load_frag16(&Ps[wave][rsel][koff]);

        // ---- PV: acc += P(16x32) * V(32x64), V fragments from LDS ----
#pragma unroll
        for (int nt = 0; nt < 4; ++nt) {
            v16bf vf = load_frag16(&Vt[buf][nt * 16 + rsel][koff]);
            acc[nt] = wmma_bf16(pf, vf, acc[nt]);
        }

        __syncthreads();   // compute on buf done before it is refilled
#if !HAVE_TDM
        if (i + 1 < NSTEP) stage(buf ^ 1, kv + 32);
#endif
    }

    // ---- normalize and write [B,S,D] bf16 (consumed by the bf16 O-proj GEMM) ----
#pragma unroll
    for (int nt = 0; nt < 4; ++nt)
#pragma unroll
        for (int r = 0; r < 8; ++r) {
            const int s = q0 + rowadd + r;
            const int d = h * DKH + nt * 16 + rsel;
            outp[((size_t)b * SEQ + s) * D_MODEL + d] = (__bf16)(acc[nt][r] / l_r[r]);
        }
}

extern "C" void kernel_launch(void* const* d_in, const int* in_sizes, int n_in,
                              void* d_out, int out_size, void* d_ws, size_t ws_size,
                              hipStream_t stream)
{
    const float* Q  = (const float*)d_in[0];
    const float* K  = (const float*)d_in[1];
    const float* V  = (const float*)d_in[2];
    const float* Wq = (const float*)d_in[3];
    const float* bq = (const float*)d_in[4];
    const float* Wk = (const float*)d_in[5];
    const float* bk = (const float*)d_in[6];
    const float* Wv = (const float*)d_in[7];
    const float* bv = (const float*)d_in[8];
    const float* Wo = (const float*)d_in[9];
    const float* bo = (const float*)d_in[10];

    char* ws = (char*)d_ws;
    const size_t nact = (size_t)M_TOT * D_MODEL * 2;     // 16 MB bf16 activation tensor
    const size_t nwt  = (size_t)D_MODEL * D_MODEL * 2;   // 2 MB bf16 weight matrix
    __bf16* Q16  = (__bf16*)(ws);
    __bf16* K16  = (__bf16*)(ws + nact);
    __bf16* V16  = (__bf16*)(ws + 2 * nact);
    __bf16* Wq16 = (__bf16*)(ws + 3 * nact);
    __bf16* Wk16 = (__bf16*)(ws + 3 * nact + nwt);
    __bf16* Wv16 = (__bf16*)(ws + 3 * nact + 2 * nwt);
    __bf16* Wo16 = (__bf16*)(ws + 3 * nact + 3 * nwt);
    __bf16* qb   = (__bf16*)(ws + 3 * nact + 4 * nwt);
    __bf16* kb2  = (__bf16*)(ws + 4 * nact + 4 * nwt);
    __bf16* vtb  = (__bf16*)(ws + 5 * nact + 4 * nwt);
    __bf16* ao   = (__bf16*)(ws + 6 * nact + 4 * nwt);

    // One-time fp32 -> bf16 conversions (activations + weights)
    const int na4 = M_TOT * D_MODEL / 4;     // 2M vec4
    const int nw4 = D_MODEL * D_MODEL / 4;   // 256K vec4
    cvt_f32_bf16<<<(na4 + 255) / 256, 256, 0, stream>>>(Q,  Q16,  na4);
    cvt_f32_bf16<<<(na4 + 255) / 256, 256, 0, stream>>>(K,  K16,  na4);
    cvt_f32_bf16<<<(na4 + 255) / 256, 256, 0, stream>>>(V,  V16,  na4);
    cvt_f32_bf16<<<(nw4 + 255) / 256, 256, 0, stream>>>(Wq, Wq16, nw4);
    cvt_f32_bf16<<<(nw4 + 255) / 256, 256, 0, stream>>>(Wk, Wk16, nw4);
    cvt_f32_bf16<<<(nw4 + 255) / 256, 256, 0, stream>>>(Wv, Wv16, nw4);
    cvt_f32_bf16<<<(nw4 + 255) / 256, 256, 0, stream>>>(Wo, Wo16, nw4);

    dim3 gg(D_MODEL / TN, M_TOT / TM);                // (8, 64)
    gemm_wmma<0><<<gg, 256, 0, stream>>>(Q16, Wq16, bq, qb);
    gemm_wmma<0><<<gg, 256, 0, stream>>>(K16, Wk16, bk, kb2);
    gemm_wmma<2><<<gg, 256, 0, stream>>>(V16, Wv16, bv, vtb);

    flash_attn<<<dim3(SEQ / 64, BATCH * HEADS), 128, 0, stream>>>(qb, kb2, vtb, ao);

    gemm_wmma<3><<<gg, 256, 0, stream>>>(ao, Wo16, bo, (float*)d_out);
}
